// EDN_91130616087314
// MI455X (gfx1250) — matrix operand order repaired and checked by reading, not compiled, not run on hardware
//
#include <hip/hip_runtime.h>
#include <math.h>

typedef _Float16 v16h __attribute__((ext_vector_type(16)));
typedef _Float16 h8   __attribute__((ext_vector_type(8)));
typedef float    v8f  __attribute__((ext_vector_type(8)));

__device__ __forceinline__ h8 ldsh8(const _Float16* p) { return *(const h8*)p; }

__device__ __forceinline__ v8f wmma16(h8 al, h8 ah, h8 bl, h8 bh, v8f c) {
  v16h a = __builtin_shufflevector(al, ah, 0,1,2,3,4,5,6,7,8,9,10,11,12,13,14,15);
  v16h b = __builtin_shufflevector(bl, bh, 0,1,2,3,4,5,6,7,8,9,10,11,12,13,14,15);
  return __builtin_amdgcn_wmma_f32_16x16x32_f16(false, a, false, b, (short)0, c, false, false);
}

// ---------------------------------------------------------------------------
// Kernel 1: scene CNN. One workgroup (256 threads, 8 waves) per sample.
// conv1 scalar fp32 -> f16 im2col in LDS -> conv2 via v_wmma_f32_16x16x32_f16
// -> spatial mean -> fc(32->128). Writes feats[b][128:256].
// Dynamic LDS layout (bytes):
//   sA   [256][160] f16   @      0   (81920)  im2col, K padded 144->160
//   sW   [ 32][160] f16   @  81920   (10240)  conv2 weights, N-major
//   sScn [64][64]  f32    @  92160   (16384)  scene; reused as conv2-out f16
//   sPrt [256]     f32    @ 108544   ( 1024)
//   sMean[32]      f32    @ 109568   (  128)
// ---------------------------------------------------------------------------
#define K1_SMEM 109696

__global__ void __launch_bounds__(256) k1_cnn(
    const float* __restrict__ g_scene,
    const float* __restrict__ c1_w, const float* __restrict__ c1_b,
    const float* __restrict__ c2_w, const float* __restrict__ c2_b,
    const float* __restrict__ fc_w, const float* __restrict__ fc_b,
    float* __restrict__ feats)
{
  extern __shared__ char smem[];
  _Float16* sA   = (_Float16*)(smem);
  _Float16* sW   = (_Float16*)(smem + 81920);
  float*    sScn = (float*)   (smem + 92160);
  float*    sPrt = (float*)   (smem + 108544);
  float*    sMean= (float*)   (smem + 109568);

  const int tid = threadIdx.x;
  const int b   = blockIdx.x;
  const float* sc = g_scene + (size_t)b * 4096;
  __builtin_prefetch(sc, 0, 0);

  // zero im2col (covers K-pad 144..159 and out-of-bounds taps)
  {
    float4 z = {0.f, 0.f, 0.f, 0.f};
    float4* a4 = (float4*)sA;
    for (int e = tid; e < (256*160*2)/16; e += 256) a4[e] = z;
  }
  // load scene (coalesced b128)
  {
    const float4* s4 = (const float4*)sc;
    float4* d4 = (float4*)sScn;
    for (int e = tid; e < 1024; e += 256) d4[e] = s4[e];
  }
  // stage conv2 weights transposed [c][t], t = ic*9+ky*3+kx, padded to 160
  for (int e = tid; e < 32*160; e += 256) {
    int c = e / 160, t = e % 160;
    sW[c*160 + t] = (t < 144) ? (_Float16)c2_w[c*144 + t] : (_Float16)0.f;
  }
  __syncthreads();

  // conv1 (1->16 ch, 3x3, stride 2, SAME: pad_before=0) + scatter into im2col
  for (int it = 0; it < 64; ++it) {
    int idx = tid + it * 256;          // 0..16383
    int ic = idx >> 10;
    int y  = (idx >> 5) & 31;
    int x  = idx & 31;
    float acc = c1_b[ic];
#pragma unroll
    for (int ky = 0; ky < 3; ++ky) {
      int iy = 2*y + ky;
#pragma unroll
      for (int kx = 0; kx < 3; ++kx) {
        int ix = 2*x + kx;
        float v = (iy < 64 && ix < 64) ? sScn[iy*64 + ix] : 0.f;
        acc = fmaf(v, c1_w[ic*9 + ky*3 + kx], acc);
      }
    }
    acc = fmaxf(acc, 0.f);
    _Float16 hv = (_Float16)acc;
    // A(m=(oy,ox), t=ic*9+ky*3+kx) = conv1out[ic][2oy+ky][2ox+kx]
    int kyb = y & 1, kxb = x & 1;
#pragma unroll
    for (int ay = 0; ay < 2; ++ay) {
      int ky = kyb + 2*ay;
      if (ky > 2 || y - ky < 0) continue;
      int oy = (y - ky) >> 1;
      if (oy >= 16) continue;
#pragma unroll
      for (int ax = 0; ax < 2; ++ax) {
        int kx = kxb + 2*ax;
        if (kx > 2 || x - kx < 0) continue;
        int ox = (x - kx) >> 1;
        if (ox >= 16) continue;
        sA[(oy*16 + ox)*160 + ic*9 + ky*3 + kx] = hv;
      }
    }
  }
  __syncthreads();

  // conv2 as WMMA: M=256, N=32, K=160(padded). 8 waves x 4 (Mt,Nt) pairs.
  const int lane = tid & 31, wv = tid >> 5;
  const int r = lane & 15, hf = lane >> 4;
  _Float16* c2o = (_Float16*)sScn;   // reuse scene region: [256][32] f16
  for (int pi = 0; pi < 4; ++pi) {
    int pair = wv*4 + pi;
    int mt = pair >> 1, nt = pair & 1;
    const _Float16* arow = sA + (mt*16 + r)*160;
    const _Float16* brow = sW + (nt*16 + r)*160;
    v8f acc = {};
#pragma unroll
    for (int s = 0; s < 5; ++s) {
      h8 al = ldsh8(arow + s*32 + 16*hf);
      h8 ah = ldsh8(arow + s*32 + 16*hf + 8);
      h8 bl = ldsh8(brow + s*32 + 16*hf);
      h8 bh = ldsh8(brow + s*32 + 16*hf + 8);
      acc = wmma16(al, ah, bl, bh, acc);
    }
    float bias = c2_b[nt*16 + r];
#pragma unroll
    for (int i = 0; i < 8; ++i) {
      float v = fmaxf(acc[i] + bias, 0.f);
      c2o[(mt*16 + i + 8*hf)*32 + nt*16 + r] = (_Float16)v;
    }
  }
  __syncthreads();

  // spatial mean over 256 positions
  {
    int c = tid & 31, g = tid >> 5;
    float s = 0.f;
    for (int m = g*32; m < g*32 + 32; ++m) s += (float)c2o[m*32 + c];
    sPrt[tid] = s;
  }
  __syncthreads();
  if (tid < 32) {
    float s = 0.f;
    for (int g = 0; g < 8; ++g) s += sPrt[g*32 + tid];
    sMean[tid] = s * (1.0f / 256.0f);
  }
  __syncthreads();
  if (tid < 128) {
    float s = fc_b[tid];
    for (int c = 0; c < 32; ++c) s = fmaf(sMean[c], fc_w[c*128 + tid], s);
    feats[(size_t)b*256 + 128 + tid] = s;
  }
}

// ---------------------------------------------------------------------------
// Kernel 2: trajectory + neighbor embeddings, one thread per sample (fp32).
// Writes feats[b][0:64] (traj) and feats[b][64:128] (vehs).
// ---------------------------------------------------------------------------
template<int IN, int OUT>
__device__ __forceinline__ void lin_relu(const float* x, const float* W, const float* Bi,
                                         float* h, bool relu) {
  for (int j = 0; j < OUT; ++j) {
    float s = Bi[j];
    for (int i = 0; i < IN; ++i) s = fmaf(x[i], W[i*OUT + j], s);
    h[j] = relu ? fmaxf(s, 0.f) : s;
  }
}

__device__ __forceinline__ void cpy(float* dst, const float* src, int n, int tid) {
  for (int e = tid; e < n; e += 256) dst[e] = src[e];
}

__global__ void __launch_bounds__(256) k2_embed(
    const float* __restrict__ obs, const float* __restrict__ sr_, const float* __restrict__ ps_,
    const float* __restrict__ et_w1, const float* __restrict__ et_b1,
    const float* __restrict__ et_w2, const float* __restrict__ et_b2,
    const float* __restrict__ et_w3, const float* __restrict__ et_b3,
    const float* __restrict__ ev_w1, const float* __restrict__ ev_b1,
    const float* __restrict__ ev_w2, const float* __restrict__ ev_b2,
    const float* __restrict__ ev_w3, const float* __restrict__ ev_b3,
    float* __restrict__ feats, int B)
{
  __shared__ float sw[7488];
  const int tid = threadIdx.x;
  cpy(sw +    0, et_w1,  512, tid);  cpy(sw +  512, et_b1,   32, tid);
  cpy(sw +  544, et_w2, 1024, tid);  cpy(sw + 1568, et_b2,   32, tid);
  cpy(sw + 1600, et_w3, 2048, tid);  cpy(sw + 3648, et_b3,   64, tid);
  cpy(sw + 3712, ev_w1,  576, tid);  cpy(sw + 4288, ev_b1,   32, tid);
  cpy(sw + 4320, ev_w2, 1024, tid);  cpy(sw + 5344, ev_b2,   32, tid);
  cpy(sw + 5376, ev_w3, 2048, tid);  cpy(sw + 7424, ev_b3,   64, tid);
  __syncthreads();

  const int b = blockIdx.x * 256 + tid;
  if (b >= B) return;
  const float* o = obs + (size_t)b * 144;   // [9][8][2]
  float sr = sr_[b], ps = ps_[b];
  float scl = sr / ps;

  // trajectory velocities -> 16 inputs
  float x[18];
  for (int t = 0; t < 8; ++t) {
    x[2*t + 0] = (o[(t+1)*16 + 0] - o[t*16 + 0] + 18.0f) * scl;
    x[2*t + 1] = (o[(t+1)*16 + 1] - o[t*16 + 1]) * scl;
  }
  float h1[32], h2[32], out64[64];
  lin_relu<16,32>(x,  sw +    0, sw +  512, h1, true);
  lin_relu<32,32>(h1, sw +  544, sw + 1568, h2, true);
  lin_relu<32,64>(h2, sw + 1600, sw + 3648, out64, false);
  for (int j = 0; j < 64; ++j) feats[(size_t)b*256 + j] = out64[j];

  // neighbor distances at last timestep, with "behind"/NaN masking
  float x0r = o[8*16 + 0];
  float d[8];
  float q0x, q0y;
  {
    float px = o[8*16 + 0], py = o[8*16 + 1];
    bool behind = !(x0r >= px);
    q0x = behind ? -1000.f : (px != px ? -1000.f : px);
    q0y = behind ? -1000.f : (py != py ? -1000.f : py);
  }
  for (int j = 0; j < 8; ++j) {
    float px = o[8*16 + j*2 + 0], py = o[8*16 + j*2 + 1];
    bool behind = !(x0r >= px);
    float qx = behind ? -1000.f : (px != px ? -1000.f : px);
    float qy = behind ? -1000.f : (py != py ? -1000.f : py);
    float dx = qx - q0x, dy = qy - q0y;
    d[j] = dx*dx + dy*dy;
  }
  // stable top-4 (ascending dist, lowest index on ties), skip self (rank 0)
  int sel[4]; unsigned used = 0;
  for (int p = 0; p < 4; ++p) {
    int best = 0; float bd = 3.4e38f;
    for (int j = 0; j < 8; ++j)
      if (!((used >> j) & 1u) && d[j] < bd) { bd = d[j]; best = j; }
    sel[p] = best; used |= 1u << best;
  }
  // relative positions for 3 closest over last 3 timesteps
  float p0x = o[8*16 + 0], p0y = o[8*16 + 1];
  float rel[3][3][2];
  for (int p = 0; p < 3; ++p) {
    int j = sel[p + 1];
    float sum = 0.f;
    for (int ti = 0; ti < 3; ++ti) {
      float rx = p0x - o[(6 + ti)*16 + j*2 + 0];
      float ry = p0y - o[(6 + ti)*16 + j*2 + 1];
      rel[ti][p][0] = rx; rel[ti][p][1] = ry;
      sum += rx + ry;
    }
    bool bad = (d[j] > 600000.0f) || (sum != sum);
    if (bad)
      for (int ti = 0; ti < 3; ++ti) { rel[ti][p][0] = 2000.f; rel[ti][p][1] = 2000.f; }
  }
  for (int ti = 0; ti < 3; ++ti)
    for (int p = 0; p < 3; ++p) {
      x[(ti*3 + p)*2 + 0] = rel[ti][p][0] * scl - 10.0f;
      x[(ti*3 + p)*2 + 1] = rel[ti][p][1] * scl - 7.0f;
    }
  lin_relu<18,32>(x,  sw + 3712, sw + 4288, h1, true);
  lin_relu<32,32>(h1, sw + 4320, sw + 5344, h2, true);
  lin_relu<32,64>(h2, sw + 5376, sw + 7424, out64, false);
  for (int j = 0; j < 64; ++j) feats[(size_t)b*256 + 64 + j] = out64[j];
}

// ---------------------------------------------------------------------------
// Kernel 3: decoder head. 16-sample tile per workgroup, 8 waves.
// feats(16x256) @ dec(256x128) -> relu -> rm1/rv1 (wave groups) -> rm2/rv2
// -> epilogue (exp/tanh/sqrt/cumsum + ps/sr scaling) -> 4 outputs.
// ---------------------------------------------------------------------------
__global__ void __launch_bounds__(256) k3_head(
    const float* __restrict__ feats, const float* __restrict__ obs,
    const float* __restrict__ sr_, const float* __restrict__ ps_,
    const float* __restrict__ dec_w, const float* __restrict__ dec_b,
    const float* __restrict__ rm_w1, const float* __restrict__ rm_b1,
    const float* __restrict__ rm_w2, const float* __restrict__ rm_b2,
    const float* __restrict__ rv_w1, const float* __restrict__ rv_b1,
    const float* __restrict__ rv_w2, const float* __restrict__ rv_b2,
    float* __restrict__ out, int B)
{
  __shared__ _Float16 sF[16*256];   // feats tile (A for dec)
  __shared__ _Float16 sB0[128*32];  // weight chunk, "mean" path / dec
  __shared__ _Float16 sB1[128*32];  // weight chunk, "var" path
  __shared__ _Float16 sH [16*128];  // features_2
  __shared__ _Float16 sHm[16*128];  // relu(rm1)
  __shared__ _Float16 sHv[16*128];  // relu(rv1)
  __shared__ float    sOm[16*24];   // pv_mean raw
  __shared__ float    sOv[16*36];   // pv_var raw

  const int tid = threadIdx.x;
  const int b0  = blockIdx.x * 16;
  const int lane = tid & 31, wv = tid >> 5;
  const int r = lane & 15, hf = lane >> 4;

  for (int e = tid; e < 4096; e += 256) {
    int m = e >> 8, k = e & 255;
    sF[e] = (_Float16)feats[(size_t)(b0 + m)*256 + k];
  }

  // ---- dec: K=256 (8 chunks), wave wv owns N-tile wv ----
  v8f acc = {};
  for (int s = 0; s < 8; ++s) {
    __syncthreads();
    for (int e = tid; e < 4096; e += 256) {        // stage [n][k'] transposed
      int k = e >> 7, n = e & 127;
      sB0[n*32 + k] = (_Float16)dec_w[(size_t)(s*32 + k)*128 + n];
    }
    __syncthreads();
    const _Float16* arow = sF + r*256 + s*32;
    const _Float16* brow = sB0 + (wv*16 + r)*32 + 16*hf;
    acc = wmma16(ldsh8(arow + 16*hf), ldsh8(arow + 16*hf + 8),
                 ldsh8(brow), ldsh8(brow + 8), acc);
  }
  {
    float bias = dec_b[wv*16 + r];
#pragma unroll
    for (int i = 0; i < 8; ++i)
      sH[(i + 8*hf)*128 + wv*16 + r] = (_Float16)fmaxf(acc[i] + bias, 0.f);
  }

  // ---- rm1 (waves 0-3) / rv1 (waves 4-7): K=128, each wave 2 N-tiles ----
  const int grp = wv >> 2, wq = wv & 3;
  v8f a0 = {}, a1 = {};
  for (int s = 0; s < 4; ++s) {
    __syncthreads();
    if (tid < 128) {
      for (int e = tid; e < 4096; e += 128) {
        int k = e >> 7, n = e & 127;
        sB0[n*32 + k] = (_Float16)rm_w1[(size_t)(s*32 + k)*128 + n];
      }
    } else {
      for (int e = tid - 128; e < 4096; e += 128) {
        int k = e >> 7, n = e & 127;
        sB1[n*32 + k] = (_Float16)rv_w1[(size_t)(s*32 + k)*128 + n];
      }
    }
    __syncthreads();
    const _Float16* arow = sH + r*128 + s*32;
    h8 al = ldsh8(arow + 16*hf), ah = ldsh8(arow + 16*hf + 8);
    const _Float16* wsrc = grp ? sB1 : sB0;
    {
      const _Float16* brow = wsrc + ((wq*2 + 0)*16 + r)*32 + 16*hf;
      a0 = wmma16(al, ah, ldsh8(brow), ldsh8(brow + 8), a0);
    }
    {
      const _Float16* brow = wsrc + ((wq*2 + 1)*16 + r)*32 + 16*hf;
      a1 = wmma16(al, ah, ldsh8(brow), ldsh8(brow + 8), a1);
    }
  }
  {
    _Float16* dst = grp ? sHv : sHm;
    const float* bias = grp ? rv_b1 : rm_b1;
#pragma unroll
    for (int t = 0; t < 2; ++t) {
      int nt = wq*2 + t;
      float bs = bias[nt*16 + r];
      v8f& a = t ? a1 : a0;
#pragma unroll
      for (int i = 0; i < 8; ++i)
        dst[(i + 8*hf)*128 + nt*16 + r] = (_Float16)fmaxf(a[i] + bs, 0.f);
    }
  }

  // ---- rm2 (waves 0-1, N=24 pad 32) / rv2 (waves 4-6, N=36 pad 48) ----
  v8f o2 = {};
  for (int s = 0; s < 4; ++s) {
    __syncthreads();
    if (tid < 128) {
      for (int e = tid; e < 32*32; e += 128) {
        int k = e >> 5, n = e & 31;
        sB0[n*32 + k] = (n < 24) ? (_Float16)rm_w2[(s*32 + k)*24 + n] : (_Float16)0.f;
      }
    } else {
      for (int e = tid - 128; e < 48*32; e += 128) {
        int k = e / 48, n = e % 48;
        sB1[n*32 + k] = (n < 36) ? (_Float16)rv_w2[(s*32 + k)*36 + n] : (_Float16)0.f;
      }
    }
    __syncthreads();
    if (wv < 2) {
      const _Float16* arow = sHm + r*128 + s*32;
      const _Float16* brow = sB0 + (wv*16 + r)*32 + 16*hf;
      o2 = wmma16(ldsh8(arow + 16*hf), ldsh8(arow + 16*hf + 8),
                  ldsh8(brow), ldsh8(brow + 8), o2);
    } else if (wv >= 4 && wv < 7) {
      const _Float16* arow = sHv + r*128 + s*32;
      const _Float16* brow = sB1 + ((wv - 4)*16 + r)*32 + 16*hf;
      o2 = wmma16(ldsh8(arow + 16*hf), ldsh8(arow + 16*hf + 8),
                  ldsh8(brow), ldsh8(brow + 8), o2);
    }
  }
  if (wv < 2) {
    int c = wv*16 + r;
    if (c < 24) {
      float bs = rm_b2[c];
#pragma unroll
      for (int i = 0; i < 8; ++i) sOm[(i + 8*hf)*24 + c] = o2[i] + bs;
    }
  } else if (wv >= 4 && wv < 7) {
    int c = (wv - 4)*16 + r;
    if (c < 36) {
      float bs = rv_b2[c];
#pragma unroll
      for (int i = 0; i < 8; ++i) sOv[(i + 8*hf)*36 + c] = o2[i] + bs;
    }
  }
  __syncthreads();

  // ---- epilogue: one thread per sample ----
  if (tid < 16) {
    int b = b0 + tid;
    float ps = ps_[b], sr = sr_[b];
    float k = ps / sr;
    float p0x = obs[(size_t)b*144 + 8*16 + 0];
    float p0y = obs[(size_t)b*144 + 8*16 + 1];
    float cx = p0x, cy = p0y;
    size_t posB = (size_t)B * 60, zB = (size_t)B * 120, spB = (size_t)B * 121;
    for (int p = 0; p < 12; ++p) {
      float mx = sOm[tid*24 + p*2 + 0], my = sOm[tid*24 + p*2 + 1];
      float e1 = expf(sOv[tid*36 + p*3 + 0]);
      float e2 = expf(sOv[tid*36 + p*3 + 1]);
      float th = tanhf(sOv[tid*36 + p*3 + 2]);
      float vx = mx * k, vy = my * k;
      size_t o5 = ((size_t)b*12 + p)*5;
      out[o5 + 0] = vx;     out[o5 + 1] = vy;
      out[o5 + 2] = e1 * k; out[o5 + 3] = e2 * k; out[o5 + 4] = th * k;
      cx += vx; cy += vy;
      out[posB + o5 + 0] = cx; out[posB + o5 + 1] = cy;
      out[posB + o5 + 2] = e1; out[posB + o5 + 3] = e2; out[posB + o5 + 4] = th;
      out[spB + (size_t)b*12 + p] = sqrtf(mx*mx + my*my) * k;
    }
    out[zB + b] = 0.f;
  }
}

// ---------------------------------------------------------------------------
extern "C" void kernel_launch(void* const* d_in, const int* in_sizes, int n_in,
                              void* d_out, int out_size, void* d_ws, size_t ws_size,
                              hipStream_t stream) {
  const float* obs   = (const float*)d_in[0];
  const float* sr    = (const float*)d_in[1];
  const float* ps    = (const float*)d_in[2];
  const float* scene = (const float*)d_in[3];
  const float* et_w1 = (const float*)d_in[4];  const float* et_b1 = (const float*)d_in[5];
  const float* et_w2 = (const float*)d_in[6];  const float* et_b2 = (const float*)d_in[7];
  const float* et_w3 = (const float*)d_in[8];  const float* et_b3 = (const float*)d_in[9];
  const float* ev_w1 = (const float*)d_in[10]; const float* ev_b1 = (const float*)d_in[11];
  const float* ev_w2 = (const float*)d_in[12]; const float* ev_b2 = (const float*)d_in[13];
  const float* ev_w3 = (const float*)d_in[14]; const float* ev_b3 = (const float*)d_in[15];
  const float* dec_w = (const float*)d_in[16]; const float* dec_b = (const float*)d_in[17];
  const float* rm_w1 = (const float*)d_in[18]; const float* rm_b1 = (const float*)d_in[19];
  const float* rm_w2 = (const float*)d_in[20]; const float* rm_b2 = (const float*)d_in[21];
  const float* rv_w1 = (const float*)d_in[22]; const float* rv_b1 = (const float*)d_in[23];
  const float* rv_w2 = (const float*)d_in[24]; const float* rv_b2 = (const float*)d_in[25];
  const float* c1_w  = (const float*)d_in[26]; const float* c1_b  = (const float*)d_in[27];
  const float* c2_w  = (const float*)d_in[28]; const float* c2_b  = (const float*)d_in[29];
  const float* fc_w  = (const float*)d_in[30]; const float* fc_b  = (const float*)d_in[31];

  const int B = in_sizes[1];                 // 8192
  float* feats = (float*)d_ws;               // B x 256 fp32 scratch

  k1_cnn<<<B, 256, K1_SMEM, stream>>>(scene, c1_w, c1_b, c2_w, c2_b, fc_w, fc_b, feats);
  k2_embed<<<(B + 255) / 256, 256, 0, stream>>>(obs, sr, ps,
      et_w1, et_b1, et_w2, et_b2, et_w3, et_b3,
      ev_w1, ev_b1, ev_w2, ev_b2, ev_w3, ev_b3, feats, B);
  k3_head<<<B / 16, 256, 0, stream>>>(feats, obs, sr, ps,
      dec_w, dec_b, rm_w1, rm_b1, rm_w2, rm_b2,
      rv_w1, rv_b1, rv_w2, rv_b2, (float*)d_out, B);
}
